// Net_2946347565967
// MI455X (gfx1250) — matrix operand order repaired and checked by reading, not compiled, not run on hardware
//
#include <hip/hip_runtime.h>

// ---------------------------------------------------------------------------
// CDNA5 (gfx1250) implementation.
// Dense layers: v_wmma_f32_16x16x32_f16, wave32, one wave per 16x16 C tile.
// GEMM operands pre-converted to f16 into K-padded (mult 32) / N-padded
// (mult 32) zero-filled buffers -> inner loop is pure b128 loads + wmma.
// Conv branch: conv+relu+maxpool fused (recompute-in-register) to avoid
// materializing pre-pool activations; saves ~1 GB of HBM traffic.
// ---------------------------------------------------------------------------

typedef __attribute__((ext_vector_type(16))) _Float16 v16h;
typedef __attribute__((ext_vector_type(8)))  _Float16 v8h;
typedef __attribute__((ext_vector_type(8)))  float    v8f;

// out[M, NP](f16) = act( A[M, KP](f16) @ W[NP, KP](f16)^T + bias[N] )
// KP, NP multiples of 32; padded regions of A/W are zero.
// Per CDNA5 ISA 7.12.2 (wave32), with h = lane>>4, l16 = lane&15:
//   A frag: row m0+l16, a[0..7]  = K[k0+8h .. +7], a[8..15] = K[k0+16+8h .. +7]
//   B frag: col n0+l16, b[0..7]  = K[k0+8h .. +7], b[8..15] = K[k0+16+8h .. +7]
//   D:      VGPR r -> M = m0 + r + 8h, N = n0 + l16.
__global__ void gemm_bias_act_wmma(const _Float16* __restrict__ A,
                                   const _Float16* __restrict__ W,
                                   const float* __restrict__ bias,
                                   _Float16* __restrict__ Out,
                                   int M, int N, int KP, int NP, int relu)
{
    const int lane = threadIdx.x & 31;
    const int wave = threadIdx.x >> 5;
    const int h    = lane >> 4;
    const int l16  = lane & 15;

    const int ntn    = NP >> 4;
    const int tile_n = blockIdx.x * (blockDim.x >> 5) + wave;   // wave-uniform
    if (tile_n >= ntn) return;                                  // uniform exit
    const int n0 = tile_n * 16;
    const int m0 = blockIdx.y * 16;                             // M % 16 == 0

    const _Float16* __restrict__ Ap = A + (size_t)(m0 + l16) * KP;
    const _Float16* __restrict__ Wp = W + (size_t)(n0 + l16) * KP;

    v8f acc = {};

    for (int k0 = 0; k0 < KP; k0 += 32) {
        __builtin_prefetch(Ap + k0 + 256, 0, 3);   // global_prefetch_b8, near
        __builtin_prefetch(Wp + k0 + 256, 0, 3);

        const v8h  alo = *(const v8h*)(Ap + k0 + 8 * h);        // 16B aligned
        const v8h  ahi = *(const v8h*)(Ap + k0 + 16 + 8 * h);   // 16B aligned
        const v16h a   = __builtin_shufflevector(alo, ahi,
                             0, 1, 2, 3, 4, 5, 6, 7, 8, 9, 10, 11, 12, 13, 14, 15);
        const v8h  blo = *(const v8h*)(Wp + k0 + 8 * h);        // 16B aligned
        const v8h  bhi = *(const v8h*)(Wp + k0 + 16 + 8 * h);   // 16B aligned
        const v16h b   = __builtin_shufflevector(blo, bhi,
                             0, 1, 2, 3, 4, 5, 6, 7, 8, 9, 10, 11, 12, 13, 14, 15);

        acc = __builtin_amdgcn_wmma_f32_16x16x32_f16(
            false, a, false, b, (short)0, acc, false, false);
    }

    const int   n    = n0 + l16;
    const bool  nin  = (n < N);
    const float bval = nin ? bias[n] : 0.0f;
#pragma unroll
    for (int r = 0; r < 8; ++r) {
        const int m = m0 + r + 8 * h;
        float v = acc[r] + bval;
        if (relu) v = fmaxf(v, 0.0f);
        Out[(size_t)m * NP + n] = nin ? (_Float16)v : (_Float16)0.0f;
    }
}

// dst[rowsP, KP](f16) <- src[rows, K](f32), zero padding outside [rows, K).
__global__ void cvt_pad_f16(const float* __restrict__ src,
                            _Float16* __restrict__ dst,
                            int rows, int rowsP, int K, int KP)
{
    const size_t total = (size_t)rowsP * KP;
    size_t idx = (size_t)blockIdx.x * blockDim.x + threadIdx.x;
    if (idx >= total) return;
    const int k = (int)(idx % KP);
    const int r = (int)(idx / KP);
    const float v = (r < rows && k < K) ? src[(size_t)r * K + k] : 0.0f;
    dst[idx] = (_Float16)v;
}

// Grouped conv1d, kernel=5, pad=2, stride=1, fused ReLU.
// Layout (B, C, L, G) with g fastest.  w: (G, Cout, Cin, 5), bias: (G, Cout).
__global__ void conv1d_group_relu(const float* __restrict__ in,
                                  const float* __restrict__ w,
                                  const float* __restrict__ bias,
                                  float* __restrict__ out,
                                  int Bn, int Cin, int Cout, int L, int G)
{
    const size_t total = (size_t)Bn * Cout * L * G;
    size_t idx = (size_t)blockIdx.x * blockDim.x + threadIdx.x;
    if (idx >= total) return;

    const int g = (int)(idx % G);  size_t t = idx / G;
    const int l = (int)(t % L);    t /= L;
    const int co = (int)(t % Cout);
    const int b  = (int)(t / Cout);

    float acc = bias[g * Cout + co];
    const float* __restrict__ wp = w + (size_t)((g * Cout + co) * Cin) * 5;
    for (int ci = 0; ci < Cin; ++ci) {
        const float* __restrict__ ip = in + ((size_t)(b * Cin + ci) * L) * G + g;
#pragma unroll
        for (int k = 0; k < 5; ++k) {
            const int li = l + k - 2;
            if (li >= 0 && li < L)
                acc += ip[(size_t)li * G] * wp[ci * 5 + k];
        }
    }
    out[idx] = fmaxf(acc, 0.0f);
}

// Fused: out[b,co,lo,g] = max_{j=0..9} relu( conv5(in)[b,co,3*lo+j,g] + bias )
// (maxpool window 10, stride 3, VALID -> 3*lo+9 < Lin always).
// Recomputes the 10 window positions in registers instead of materializing
// the full-length conv output (saves a 2x full-tensor HBM round trip).
__global__ void conv_relu_maxpool(const float* __restrict__ in,
                                  const float* __restrict__ w,
                                  const float* __restrict__ bias,
                                  float* __restrict__ out,
                                  int Bn, int Cin, int Cout, int Lin, int Lout, int G)
{
    const size_t total = (size_t)Bn * Cout * Lout * G;
    size_t idx = (size_t)blockIdx.x * blockDim.x + threadIdx.x;
    if (idx >= total) return;

    const int g  = (int)(idx % G);  size_t t = idx / G;
    const int lo = (int)(t % Lout); t /= Lout;
    const int co = (int)(t % Cout);
    const int b  = (int)(t / Cout);

    const float bv = bias[g * Cout + co];
    const float* __restrict__ wp = w + (size_t)((g * Cout + co) * Cin) * 5;

    float m = 0.0f;   // relu output is >= 0, so 0 is a safe identity for max
#pragma unroll
    for (int j = 0; j < 10; ++j) {
        const int l = 3 * lo + j;       // < Lin (valid pooling)
        float acc = bv;
        for (int ci = 0; ci < Cin; ++ci) {
            const float* __restrict__ ip =
                in + ((size_t)(b * Cin + ci) * Lin) * G + g;
#pragma unroll
            for (int k = 0; k < 5; ++k) {
                const int li = l + k - 2;
                if (li >= 0 && li < Lin)
                    acc += ip[(size_t)li * G] * wp[ci * 5 + k];
            }
        }
        m = fmaxf(m, acc);              // max(relu(a)) == relu(max(a)), a>=0 base
    }
    out[idx] = m;
}

// Reference "mlp": reshape (B,C,L,F)->(B*L,C,F) is a raw reinterpretation, so
// the weight channel index is p % C where p enumerates (b,c,l) row-major.
__global__ void mlp_relu(const float* __restrict__ in,
                         const float* __restrict__ w,
                         const float* __restrict__ bias,
                         float* __restrict__ out,
                         size_t P, int C, int Fin, int Fout)
{
    const size_t total = P * (size_t)Fout;
    size_t idx = (size_t)blockIdx.x * blockDim.x + threadIdx.x;
    if (idx >= total) return;

    const int    o = (int)(idx % Fout);
    const size_t p = idx / Fout;
    const int    i = (int)(p % C);

    float acc = bias[i * Fout + o];
    const float* __restrict__ wp = w + (size_t)(i * Fout + o) * Fin;
    const float* __restrict__ ip = in + p * Fin;
    for (int f = 0; f < Fin; ++f) acc += ip[f] * wp[f];
    out[idx] = fmaxf(acc, 0.0f);
}

// zh[b, 0:30] = enc1(f32), zh[b, 30:130] = e2h (stride 128), zh[b,130:160]=0.
__global__ void concat_enc_f16(const float* __restrict__ enc1,
                               const _Float16* __restrict__ e2h,
                               _Float16* __restrict__ zh, int Bn)
{
    const size_t total = (size_t)Bn * 160;
    size_t idx = (size_t)blockIdx.x * blockDim.x + threadIdx.x;
    if (idx >= total) return;
    const int j = (int)(idx % 160);
    const int b = (int)(idx / 160);
    _Float16 v = (_Float16)0.0f;
    if (j < 30)       v = (_Float16)enc1[(size_t)b * 30 + j];
    else if (j < 130) v = e2h[(size_t)b * 128 + (j - 30)];
    zh[idx] = v;
}

// out[b] = z3h[b, 0:20] . last_w + last_b   (z3h stride 32, f16)
__global__ void final_linear_f16(const _Float16* __restrict__ z3h,
                                 const float* __restrict__ w,
                                 const float* __restrict__ b,
                                 float* __restrict__ out, int Bn)
{
    int bi = blockIdx.x * blockDim.x + threadIdx.x;
    if (bi >= Bn) return;
    float acc = b[0];
#pragma unroll
    for (int j = 0; j < 20; ++j) acc += (float)z3h[(size_t)bi * 32 + j] * w[j];
    out[bi] = acc;
}

// ---------------------------------------------------------------------------

extern "C" void kernel_launch(void* const* d_in, const int* in_sizes, int n_in,
                              void* d_out, int out_size, void* d_ws, size_t ws_size,
                              hipStream_t stream)
{
    (void)in_sizes; (void)n_in; (void)out_size; (void)ws_size;

    const int Bn = 1024, T = 482, G1 = 55, G2 = 10;

    const float* x  = (const float*)d_in[0];
    const float* x2 = (const float*)d_in[1];
    const float* cw[8], * cb[8];
    for (int i = 0; i < 8; ++i) {
        cw[i] = (const float*)d_in[2 + 2 * i];
        cb[i] = (const float*)d_in[3 + 2 * i];
    }
    const float* lin1_w = (const float*)d_in[18];
    const float* lin1_b = (const float*)d_in[19];
    const float* lin2_w = (const float*)d_in[20];
    const float* lin2_b = (const float*)d_in[21];
    const float* l1_w = (const float*)d_in[22]; const float* l1_b = (const float*)d_in[23];
    const float* l2_w = (const float*)d_in[24]; const float* l2_b = (const float*)d_in[25];
    const float* l3_w = (const float*)d_in[26]; const float* l3_b = (const float*)d_in[27];
    const float* c1_w = (const float*)d_in[28]; const float* c1_b = (const float*)d_in[29];
    const float* c2_w = (const float*)d_in[30]; const float* c2_b = (const float*)d_in[31];
    const float* c3_w = (const float*)d_in[32]; const float* c3_b = (const float*)d_in[33];
    const float* lw   = (const float*)d_in[34]; const float* lb   = (const float*)d_in[35];
    float* out = (float*)d_out;

    // --- workspace: two big f32 ping-pong conv buffers, then f16 region ---
    float* ws = (float*)d_ws;
    const size_t nBig = (size_t)Bn * 3 * T * G1;     // 81,438,720 floats each
    float* bufA = ws;
    float* bufB = ws + nBig;
    _Float16* hp = (_Float16*)(ws + 2 * nBig);       // 64B-aligned region
    auto carve = [&](size_t nhalfs) { _Float16* p = hp; hp += nhalfs; return p; };

    // padded dims (KP/NP multiples of 32)
    const int KP1 = 1312, NP1 = 704;   // l1: K=1307, N=700
    const int KP2 = 704,  NP2 = 224;   // l2: K=700,  N=200
    const int KP3 = 224,  NP3 = 128;   // l3: K=200,  N=100
    const int KPc1 = 160, NPc1 = 128;  // c1: K=130,  N=100
    const int KPc2 = 128, NPc2 = 128;  // c2: K=100,  N=100
    const int KPc3 = 128, NPc3 = 32;   // c3: K=100,  N=20

    _Float16* X2h = carve((size_t)Bn * KP1);
    _Float16* W1h = carve((size_t)NP1 * KP1);
    _Float16* Y1h = carve((size_t)Bn * NP1);
    _Float16* W2h = carve((size_t)NP2 * KP2);
    _Float16* Y2h = carve((size_t)Bn * NP2);
    _Float16* W3h = carve((size_t)NP3 * KP3);
    _Float16* E2h = carve((size_t)Bn * NP3);
    _Float16* Zh  = carve((size_t)Bn * KPc1);
    _Float16* C1h = carve((size_t)NPc1 * KPc1);
    _Float16* Z1h = carve((size_t)Bn * NPc1);
    _Float16* C2h = carve((size_t)NPc2 * KPc2);
    _Float16* Z2h = carve((size_t)Bn * NPc2);
    _Float16* C3h = carve((size_t)NPc3 * KPc3);
    _Float16* Z3h = carve((size_t)Bn * NPc3);

    auto cdiv = [](size_t a, size_t b) { return (unsigned)((a + b - 1) / b); };

    auto cvt = [&](const float* src, _Float16* dst, int rows, int rowsP, int K, int KP) {
        size_t total = (size_t)rowsP * KP;
        cvt_pad_f16<<<cdiv(total, 256), 256, 0, stream>>>(src, dst, rows, rowsP, K, KP);
    };
    auto conv = [&](const float* in, int ci, int L, int G, int wi, float* o) {
        size_t total = (size_t)Bn * 3 * L * G;
        conv1d_group_relu<<<cdiv(total, 256), 256, 0, stream>>>(
            in, cw[wi], cb[wi], o, Bn, ci, 3, L, G);
    };
    auto convpool = [&](const float* in, int Lin, int Lout, int G, int wi, float* o) {
        size_t total = (size_t)Bn * 3 * Lout * G;
        conv_relu_maxpool<<<cdiv(total, 256), 256, 0, stream>>>(
            in, cw[wi], cb[wi], o, Bn, 3, 3, Lin, Lout, G);
    };
    auto gemm = [&](const _Float16* A, const _Float16* W, const float* bias,
                    _Float16* O, int N, int KP, int NP, int relu) {
        int ntn = NP / 16;
        dim3 grid((unsigned)((ntn + 3) / 4), (unsigned)(Bn / 16));
        gemm_bias_act_wmma<<<grid, 128, 0, stream>>>(A, W, bias, O, Bn, N, KP, NP, relu);
    };

    // --- operand conversions (f32 -> padded f16) ---
    cvt(x2,   X2h, Bn,  Bn,   1307, KP1);
    cvt(l1_w, W1h, 700, NP1,  1307, KP1);
    cvt(l2_w, W2h, 200, NP2,  700,  KP2);
    cvt(l3_w, W3h, 100, NP3,  200,  KP3);
    cvt(c1_w, C1h, 100, NPc1, 130,  KPc1);
    cvt(c2_w, C2h, 100, NPc2, 100,  KPc2);
    cvt(c3_w, C3h, 20,  NPc3, 100,  KPc3);

    // --- conv branch: L = 482 -> 158 -> 50 -> 14 -> 2 (pools fused) ---
    conv(x, 1, T, G1, 0, bufA);                       // conv1 (Cin=1), relu
    convpool(bufA, T, 158, G1, 1, bufB);              // conv2 + relu + pool1
    conv(bufB, 3, 158, G1, 2, bufA);                  // conv3, relu
    convpool(bufA, 158, 50, G1, 3, bufB);             // conv4 + relu + pool2
    {   // mlp1: F 55 -> 10
        size_t P = (size_t)Bn * 3 * 50;
        mlp_relu<<<cdiv(P * 10, 256), 256, 0, stream>>>(
            bufB, lin1_w, lin1_b, bufA, P, 3, 55, 10);
    }
    conv(bufA, 3, 50, G2, 4, bufB);                   // conv5, relu
    convpool(bufB, 50, 14, G2, 5, bufA);              // conv6 + relu + pool3
    conv(bufA, 3, 14, G2, 6, bufB);                   // conv7, relu
    convpool(bufB, 14, 2, G2, 7, bufA);               // conv8 + relu + pool4
    {   // mlp2: F 10 -> 5  -> enc1 in bufB as (B, 30) f32
        size_t P = (size_t)Bn * 3 * 2;
        mlp_relu<<<cdiv(P * 5, 256), 256, 0, stream>>>(
            bufA, lin2_w, lin2_b, bufB, P, 3, 10, 5);
    }

    // --- dense branch (WMMA GEMMs, f16 chained) ---
    gemm(X2h, W1h, l1_b, Y1h, 700, KP1, NP1, 1);
    gemm(Y1h, W2h, l2_b, Y2h, 200, KP2, NP2, 1);
    gemm(Y2h, W3h, l3_b, E2h, 100, KP3, NP3, 0);

    // --- head ---
    concat_enc_f16<<<cdiv((size_t)Bn * 160, 256), 256, 0, stream>>>(bufB, E2h, Zh, Bn);
    gemm(Zh,  C1h, c1_b, Z1h, 100, KPc1, NPc1, 1);
    gemm(Z1h, C2h, c2_b, Z2h, 100, KPc2, NPc2, 1);
    gemm(Z2h, C3h, c3_b, Z3h, 20,  KPc3, NPc3, 1);
    final_linear_f16<<<cdiv((size_t)Bn, 256), 256, 0, stream>>>(Z3h, lw, lb, out, Bn);
}
